// SSAN_DWAP_3874060501758
// MI455X (gfx1250) — matrix operand order repaired and checked by reading, not compiled, not run on hardware
//
#include <hip/hip_runtime.h>
#include <hip/hip_bf16.h>
#include <math.h>

typedef _Float16 f16;
typedef __attribute__((ext_vector_type(16))) _Float16 v16h;
typedef __attribute__((ext_vector_type(8)))  float    v8f;
typedef __attribute__((ext_vector_type(4)))  unsigned int v4u;
typedef __attribute__((ext_vector_type(8)))  unsigned int v8u;

#define Bb   8
#define Tt   48
#define Cc   684
#define Hh   16
#define Ww   64
#define HWp  1024
#define Pn   8192
#define Ee   256
#define HD   256
#define Aa   512
#define Vv   111
#define Ll   432
#define KCPAD 704   /* 684 padded to 22*32 */
#define KTAP  121
#define KTPAD 128   /* 121 padded to 4*32 */

// ---------------------------------------------------------------------------
// Pre: f16 weight staging. cf16t[a][kpad=704] = conv_feat[a][c] (K-contiguous
// per output column -> each WMMA B-lane reads 32 contiguous bytes).
// kf16t[a][kpad=128] = Kfused[a][tap] = sum_c W_att[c][a]*conv_cov[c][tap]
// (fuses the 11x11 coverage conv with the 512x512 W_att projection).
// ---------------------------------------------------------------------------
__global__ void k_prep_weights(const float* conv_feat, const float* W_att,
                               const float* conv_cov, f16* cf16t, f16* kf16t) {
  int id = blockIdx.x * blockDim.x + threadIdx.x;
  const int N1 = Aa * KCPAD;
  if (id < N1) {
    int a = id / KCPAD, c = id % KCPAD;
    float v = (c < Cc) ? conv_feat[(size_t)a * Cc + c] : 0.f;
    cf16t[id] = (f16)v;
  } else {
    int id2 = id - N1;
    if (id2 < Aa * KTPAD) {
      int a = id2 / KTPAD, tap = id2 % KTPAD;
      float s = 0.f;
      if (tap < KTAP)
        for (int c = 0; c < Aa; ++c)
          s += W_att[(size_t)c * Aa + a] * conv_cov[(size_t)c * KTAP + tap];
      kf16t[id2] = (f16)s;
    }
  }
}

// zero alpha_sum; masked channel sums for the init-hidden average
__global__ void k_init(const float* feat, const float* imask,
                       float* alpha_sum, float* num, float* msum) {
  int id = blockIdx.x * blockDim.x + threadIdx.x;
  if (id < Pn) alpha_sum[id] = 0.f;
  if (id < Bb * Cc) {
    int b = id / Cc, c = id % Cc;
    const float* fp = feat + ((size_t)(b * Cc + c) << 10);
    const float* mp = imask + (size_t)b * 256 * 1024;
    float s = 0.f;
    for (int hw = 0; hw < HWp; ++hw)
      s += fp[hw] * mp[(hw >> 6) * 16 * 1024 + (hw & 63) * 16];
    num[id] = s;
  }
  if (id >= Bb * Cc && id < Bb * Cc + Bb) {
    int b = id - Bb * Cc;
    const float* mp = imask + (size_t)b * 256 * 1024;
    float s = 0.f;
    for (int hw = 0; hw < HWp; ++hw)
      s += mp[(hw >> 6) * 16 * 1024 + (hw & 63) * 16];
    msum[b] = s;
  }
}

__global__ void __launch_bounds__(256)
k_hidden0_loc(const float* num, const float* msum, const float* W_init,
              const float* b_init, const float* loc, const float* W_l,
              const float* b_l, float* hidden, float* locw) {
  int tid = threadIdx.x;
  for (int o = tid; o < Bb * HD; o += 256) {
    int b = o >> 8, k = o & 255;
    float inv = 1.f / msum[b];
    float s = b_init[k];
    const float* nb = num + b * Cc;
    for (int c = 0; c < Cc; ++c) s += nb[c] * inv * W_init[(size_t)c * HD + k];
    hidden[o] = tanhf(s);
    float s2 = b_l[k];
    const float* lb = loc + b * Ll;
    for (int l = 0; l < Ll; ++l) s2 += lb[l] * W_l[(size_t)l * HD + k];
    locw[o] = s2;
  }
}

// ---------------------------------------------------------------------------
// 1x1 conv as WMMA GEMM: ctp[p][a] = sum_c feat[p][c]*conv_feat[a][c] + b_feat
// A tile (32 channels x 16 positions, row stride 1024 f32) is DMA'd to LDS by
// the Tensor Data Mover; tensor_dim1 = remaining channels gives free zero-fill
// on the padded final K-chunk (TDM OOB loads return zero).
// ---------------------------------------------------------------------------
__global__ void __launch_bounds__(256)
k_cnn_trans(const float* feat, const f16* cf16t, const float* b_feat,
            float* ctp) {
  int blk = blockIdx.x;
  int mt = blk >> 2, nb = blk & 3;
  int p0 = mt * 16;
  int b = p0 >> 10, hw0 = p0 & 1023;
  int tid = threadIdx.x, lane = tid & 31, wid = tid >> 5;
  __shared__ float Alds[32 * 16];  // [channel][pos], written by TDM
  v8f acc = {};
  int aN = nb * 128 + wid * 16 + (lane & 15);
  int kb_a = (lane < 16) ? 0 : 16;
  int kb_b = (lane < 16) ? 0 : 16;
  int Ma = lane & 15;
  unsigned lds_off = (unsigned)(size_t)(&Alds[0]);  // low 32 bits = LDS offset
  for (int kc = 0; kc < 22; ++kc) {
    if (wid == 0) {
      int c0 = kc * 32;
      unsigned long long ga =
          (unsigned long long)(size_t)(feat + ((size_t)(b * Cc + c0) << 10) + hw0);
      unsigned rows = (unsigned)(Cc - c0);  // remaining rows; <32 on last chunk
      // D# group 0 (CDNA5 ISA): count=1 | lds_addr | global_addr | type=2
      v4u g0;
      g0[0] = 1u;
      g0[1] = lds_off;
      g0[2] = (unsigned)ga;
      g0[3] = ((unsigned)(ga >> 32) & 0x01FFFFFFu) | (2u << 30);
      // D# group 1 (S8.4): data_size=4B, tensor_dim0=1024, tensor_dim1=rows,
      // tile_dim0=16, tile_dim1=32, tensor_dim0_stride=1024
      v8u g1;
      g1[0] = 2u << 16;                                   // data_size code 2
      g1[1] = (1024u & 0xFFFFu) << 16;                    // tensor_dim0[15:0]
      g1[2] = (1024u >> 16) | ((rows & 0xFFFFu) << 16);   // dim0 hi | dim1 lo
      g1[3] = (rows >> 16) | (16u << 16);                 // dim1 hi | tile_dim0
      g1[4] = 32u;                                        // tile_dim1 (tile_dim2=0)
      g1[5] = 1024u;                                      // tensor_dim0_stride lo
      g1[6] = 0u;
      g1[7] = 0u;
      asm volatile("tensor_load_to_lds %0, %1" : : "s"(g0), "s"(g1) : "memory");
      __builtin_amdgcn_s_wait_tensorcnt(0);
    }
    __syncthreads();
    union { v16h v; f16 h[16]; } af, bf;
#pragma unroll
    for (int i = 0; i < 16; ++i) {
      int K = kb_a + i;   // contiguous k = 16h + i, matches B
      af.h[i] = (f16)Alds[K * 16 + Ma];
    }
    const f16* bbp = cf16t + (size_t)aN * KCPAD + kc * 32 + kb_b;
#pragma unroll
    for (int i = 0; i < 16; ++i) bf.h[i] = bbp[i];
    acc = __builtin_amdgcn_wmma_f32_16x16x32_f16(false, af.v, false, bf.v,
                                                 (short)0, acc, false, false);
    __syncthreads();
  }
  float bfa = b_feat[aN];
#pragma unroll
  for (int r = 0; r < 8; ++r) {
    int M = r + ((lane >= 16) ? 8 : 0);
    ctp[(size_t)(p0 + M) * Aa + aN] = acc[r] + bfa;
  }
}

// GRU + query, one tiny block (B*HD = 2048 outputs)
__global__ void __launch_bounds__(256)
k_gru(const float* hidden_in, const int* labels, const float* embW,
      const float* w_ih, const float* w_hh, const float* b_ih,
      const float* b_hh, const float* W_hq, const float* b_hq,
      float* hidden_out, float* embbuf, float* query, int t) {
  __shared__ float hS[Bb * HD], eS[Bb * Ee], hN[Bb * HD];
  int tid = threadIdx.x;
  for (int i = tid; i < Bb * HD; i += 256) {
    hS[i] = hidden_in[i];
    int b = i >> 8, e = i & 255;
    int id = (t == 0) ? 1 : labels[b * Tt + (t - 1)];
    eS[i] = embW[(size_t)id * Ee + e];
  }
  __syncthreads();
  for (int o = tid; o < Bb * HD; o += 256) {
    int b = o >> 8, k = o & 255;
    const float* eb = eS + b * Ee;
    const float* hb = hS + b * HD;
    float ir = b_ih[k], iz = b_ih[HD + k], inn = b_ih[2 * HD + k];
    float hr = b_hh[k], hz = b_hh[HD + k], hn = b_hh[2 * HD + k];
    const float* wr = w_ih + (size_t)k * Ee;
    const float* wz = w_ih + (size_t)(HD + k) * Ee;
    const float* wn = w_ih + (size_t)(2 * HD + k) * Ee;
    const float* vr = w_hh + (size_t)k * HD;
    const float* vz = w_hh + (size_t)(HD + k) * HD;
    const float* vn = w_hh + (size_t)(2 * HD + k) * HD;
    for (int e = 0; e < Ee; ++e) {
      float ev = eb[e], hv = hb[e];
      ir += ev * wr[e]; iz += ev * wz[e]; inn += ev * wn[e];
      hr += hv * vr[e]; hz += hv * vz[e]; hn += hv * vn[e];
    }
    float r = 1.f / (1.f + expf(-(ir + hr)));
    float z = 1.f / (1.f + expf(-(iz + hz)));
    float nc = tanhf(inn + r * hn);
    hN[o] = (1.f - z) * nc + z * hb[k];
  }
  __syncthreads();
  for (int i = tid; i < Bb * HD; i += 256) { hidden_out[i] = hN[i]; embbuf[i] = eS[i]; }
  for (int o = tid; o < Bb * Aa; o += 256) {
    int b = o >> 9, a = o & 511;
    float s = b_hq[a];
    const float* hb = hN + b * HD;
    for (int k = 0; k < HD; ++k) s += hb[k] * W_hq[(size_t)k * Aa + a];
    query[o] = s;
  }
}

// ---------------------------------------------------------------------------
// Fused coverage-conv (im2col WMMA) + tanh + energy projection.
// Block covers 2 rows x 64 cols of one batch; wave = 16 positions.
// A-fragments (taps) are N-invariant -> hoisted; 32 N-tiles x 4 K-chunks.
// ---------------------------------------------------------------------------
__global__ void __launch_bounds__(256)
k_attmap(const float* alpha_sum, const f16* kf16t, const float* query,
         const float* ctp, const float* W_e, const float* b_e, float* energy) {
  int blk = blockIdx.x;
  int b = blk >> 3;
  int h0 = (blk & 7) * 2;
  int tid = threadIdx.x, lane = tid & 31, wid = tid >> 5;
  __shared__ float alds[12 * 74];
  __shared__ float qS[Aa], weS[Aa];
  for (int i = tid; i < 12 * 74; i += 256) {
    int rr = i / 74, cc = i % 74;
    int hh = h0 + rr - 5, wc = cc - 5;
    float v = 0.f;
    if (hh >= 0 && hh < Hh && wc >= 0 && wc < Ww)
      v = alpha_sum[b * HWp + hh * Ww + wc];
    alds[i] = v;
  }
  for (int i = tid; i < Aa; i += 256) { qS[i] = query[b * Aa + i]; weS[i] = W_e[i]; }
  __syncthreads();
  int hr = wid >> 2;
  int w0 = (wid & 3) * 16;
  int Ma = lane & 15;
  int w = w0 + Ma;
  int kb_a = (lane < 16) ? 0 : 16;
  int kb_b = (lane < 16) ? 0 : 16;
  union F { v16h v; f16 h[16]; };
  F afr[4];
#pragma unroll
  for (int kc = 0; kc < 4; ++kc) {
#pragma unroll
    for (int i = 0; i < 16; ++i) {
      int K = kc * 32 + kb_a + i;
      float v = 0.f;
      if (K < KTAP) v = alds[(hr + K / 11) * 74 + (w + K % 11)];
      afr[kc].h[i] = (f16)v;
    }
  }
  int pbase = b * HWp + (h0 + hr) * Ww;
  float eacc[8];
#pragma unroll
  for (int r = 0; r < 8; ++r) eacc[r] = 0.f;
  int aN = lane & 15;
  for (int nt = 0; nt < 32; ++nt) {
    int a = nt * 16 + aN;
    v8f acc = {};
#pragma unroll
    for (int kc = 0; kc < 4; ++kc) {
      F bf;
      const f16* bbp = kf16t + (size_t)a * KTPAD + kc * 32 + kb_b;
#pragma unroll
      for (int i = 0; i < 16; ++i) bf.h[i] = bbp[i];
      acc = __builtin_amdgcn_wmma_f32_16x16x32_f16(false, afr[kc].v, false,
                                                   bf.v, (short)0, acc,
                                                   false, false);
    }
    float qv = qS[a], wev = weS[a];
#pragma unroll
    for (int r = 0; r < 8; ++r) {
      int M = r + ((lane >= 16) ? 8 : 0);
      int p = pbase + w0 + M;
      float s = tanhf(acc[r] + qv + ctp[(size_t)p * Aa + a]);
      eacc[r] += s * wev;
    }
  }
  float be = b_e[0];
#pragma unroll
  for (int r = 0; r < 8; ++r) {
    float v = eacc[r];
    v += __shfl_xor(v, 1);
    v += __shfl_xor(v, 2);
    v += __shfl_xor(v, 4);
    v += __shfl_xor(v, 8);   // reduce within each 16-lane half (same M set)
    if ((lane & 15) == 0) {
      int M = r + ((lane >= 16) ? 8 : 0);
      energy[pbase + w0 + M] = v + be;
    }
  }
}

__global__ void k_maxred(const float* energy, float* maxv) {
  __shared__ float red[256];
  int tid = threadIdx.x;
  float m = -1e30f;
  for (int i = tid; i < Pn; i += 256) m = fmaxf(m, energy[i]);
  red[tid] = m;
  __syncthreads();
  for (int s = 128; s > 0; s >>= 1) {
    if (tid < s) red[tid] = fmaxf(red[tid], red[tid + s]);
    __syncthreads();
  }
  if (tid == 0) maxv[0] = red[0];
}

__global__ void k_expsum(const float* energy, const float* maxv,
                         const float* imask, float* e_arr, float* sums) {
  int b = blockIdx.x, tid = threadIdx.x;
  __shared__ float red[256];
  float mv = maxv[0];
  float s = 0.f;
  const float* mp = imask + (size_t)b * 256 * 1024;
  for (int hw = tid; hw < HWp; hw += 256) {
    float m2 = mp[(hw >> 6) * 16 * 1024 + (hw & 63) * 16];
    float ev = expf(energy[b * HWp + hw] - mv) * m2;
    e_arr[b * HWp + hw] = ev;
    s += ev;
  }
  red[tid] = s;
  __syncthreads();
  for (int st = 128; st > 0; st >>= 1) {
    if (tid < st) red[tid] += red[tid + st];
    __syncthreads();
  }
  if (tid == 0) sums[b] = red[0];
}

__global__ void k_alpha(float* e_arr, const float* sums, float* alpha_sum,
                        float* out_alphas, int t) {
  int id = blockIdx.x * blockDim.x + threadIdx.x;
  if (id >= Pn) return;
  int b = id >> 10, hw = id & 1023;
  float alpha = e_arr[id] / (sums[b] + 1e-10f);
  out_alphas[((size_t)(b * Tt + t) << 10) + hw] = alpha;
  alpha_sum[id] += alpha;
  e_arr[id] = (alpha > 0.02f) ? alpha : 0.f;  // alpha*amask for ctx
}

__global__ void k_ctx(const float* e_arr, const float* feat, float* ctx) {
  int id = blockIdx.x * blockDim.x + threadIdx.x;
  if (id >= Bb * Cc) return;
  int b = id / Cc, c = id % Cc;
  const float* ap = e_arr + b * HWp;
  const float* fp = feat + ((size_t)(b * Cc + c) << 10);
  float s = 0.f;
  for (int hw = 0; hw < HWp; ++hw) s += ap[hw] * fp[hw];
  ctx[id] = s;
}

__global__ void __launch_bounds__(256)
k_output(const float* hidden, const float* embbuf, const float* ctx,
         const float* W_s, const float* b_s, const float* W_we,
         const float* b_we, const float* W_c, const float* b_c,
         const float* locw, const float* W_o, const float* b_o,
         float* out_probs, int t) {
  __shared__ float oS[Bb * HD];
  int tid = threadIdx.x;
  for (int o = tid; o < Bb * HD; o += 256) {
    int b = o >> 8, k = o & 255;
    float s = b_s[k] + b_we[k] + b_c[k];
    const float* hb = hidden + b * HD;
    const float* eb = embbuf + b * Ee;
    for (int e = 0; e < HD; ++e)
      s += hb[e] * W_s[(size_t)e * HD + k] + eb[e] * W_we[(size_t)e * HD + k];
    const float* cb = ctx + b * Cc;
    for (int c = 0; c < Cc; ++c) s += cb[c] * W_c[(size_t)c * HD + k];
    oS[o] = fmaxf(s, locw[o]);
  }
  __syncthreads();
  for (int o = tid; o < Bb * Vv; o += 256) {
    int b = o / Vv, v = o % Vv;
    float s = b_o[v];
    const float* ob = oS + b * HD;
    for (int k = 0; k < HD; ++k) s += ob[k] * W_o[(size_t)k * Vv + v];
    out_probs[(size_t)(b * Tt + t) * Vv + v] = s;
  }
}

extern "C" void kernel_launch(void* const* d_in, const int* in_sizes, int n_in,
                              void* d_out, int out_size, void* d_ws,
                              size_t ws_size, hipStream_t stream) {
  (void)in_sizes; (void)n_in; (void)out_size; (void)ws_size;
  const float* feat      = (const float*)d_in[0];
  const int*   labels    = (const int*)d_in[1];
  const float* loc       = (const float*)d_in[2];
  const float* imask     = (const float*)d_in[3];
  const float* W_init    = (const float*)d_in[5];
  const float* b_init    = (const float*)d_in[6];
  const float* embW      = (const float*)d_in[7];
  const float* w_ih      = (const float*)d_in[8];
  const float* w_hh      = (const float*)d_in[9];
  const float* b_ih      = (const float*)d_in[10];
  const float* b_hh      = (const float*)d_in[11];
  const float* W_hq      = (const float*)d_in[12];
  const float* b_hq      = (const float*)d_in[13];
  const float* conv_cov  = (const float*)d_in[14];
  const float* W_att     = (const float*)d_in[15];
  const float* W_e       = (const float*)d_in[16];
  const float* b_e       = (const float*)d_in[17];
  const float* conv_feat = (const float*)d_in[18];
  const float* b_feat    = (const float*)d_in[19];
  const float* W_s       = (const float*)d_in[20];
  const float* b_s       = (const float*)d_in[21];
  const float* W_we      = (const float*)d_in[22];
  const float* b_we      = (const float*)d_in[23];
  const float* W_c       = (const float*)d_in[24];
  const float* b_c       = (const float*)d_in[25];
  const float* W_l       = (const float*)d_in[26];
  const float* b_l       = (const float*)d_in[27];
  const float* W_o       = (const float*)d_in[28];
  const float* b_o       = (const float*)d_in[29];

  char* ws = (char*)d_ws;
  size_t off = 0;
  auto alloc = [&](size_t bytes) -> void* {
    void* p = ws + off;
    off = (off + bytes + 255) & ~(size_t)255;
    return p;
  };
  float* ctp     = (float*)alloc((size_t)Pn * Aa * 4);
  f16*   cf16t   = (f16*)alloc((size_t)Aa * KCPAD * 2);
  f16*   kf16t   = (f16*)alloc((size_t)Aa * KTPAD * 2);
  float* hidden  = (float*)alloc(Bb * HD * 4);
  float* embbuf  = (float*)alloc(Bb * Ee * 4);
  float* query   = (float*)alloc(Bb * Aa * 4);
  float* alpha_s = (float*)alloc(Pn * 4);
  float* energy  = (float*)alloc(Pn * 4);
  float* e_arr   = (float*)alloc(Pn * 4);
  float* sums    = (float*)alloc(Bb * 4);
  float* maxv    = (float*)alloc(4);
  float* num     = (float*)alloc(Bb * Cc * 4);
  float* msum    = (float*)alloc(Bb * 4);
  float* locw    = (float*)alloc(Bb * HD * 4);
  float* ctx     = (float*)alloc(Bb * Cc * 4);

  float* out_probs  = (float*)d_out;
  float* out_alphas = out_probs + Bb * Tt * Vv;

  k_prep_weights<<<(Aa * KCPAD + Aa * KTPAD + 255) / 256, 256, 0, stream>>>(
      conv_feat, W_att, conv_cov, cf16t, kf16t);
  k_init<<<Pn / 256, 256, 0, stream>>>(feat, imask, alpha_s, num, msum);
  k_hidden0_loc<<<1, 256, 0, stream>>>(num, msum, W_init, b_init, loc, W_l,
                                       b_l, hidden, locw);
  k_cnn_trans<<<2048, 256, 0, stream>>>(feat, cf16t, b_feat, ctp);

  for (int t = 0; t < Tt; ++t) {
    k_gru<<<1, 256, 0, stream>>>(hidden, labels, embW, w_ih, w_hh, b_ih, b_hh,
                                 W_hq, b_hq, hidden, embbuf, query, t);
    k_attmap<<<64, 256, 0, stream>>>(alpha_s, kf16t, query, ctp, W_e, b_e,
                                     energy);
    k_maxred<<<1, 256, 0, stream>>>(energy, maxv);
    k_expsum<<<Bb, 256, 0, stream>>>(energy, maxv, imask, e_arr, sums);
    k_alpha<<<Pn / 256, 256, 0, stream>>>(e_arr, sums, alpha_s, out_alphas, t);
    k_ctx<<<(Bb * Cc + 255) / 256, 256, 0, stream>>>(e_arr, feat, ctx);
    k_output<<<1, 256, 0, stream>>>(hidden, embbuf, ctx, W_s, b_s, W_we, b_we,
                                    W_c, b_c, locw, W_o, b_o, out_probs, t);
  }
}
